// NT_Xent_12137577579241
// MI455X (gfx1250) — matrix operand order repaired and checked by reading, not compiled, not run on hardware
//
#include <hip/hip_runtime.h>
#include <hip/hip_bf16.h>

typedef __attribute__((ext_vector_type(16))) _Float16 v16h;
typedef __attribute__((ext_vector_type(8)))  _Float16 v8h;
typedef __attribute__((ext_vector_type(4)))  _Float16 v4h;
typedef __attribute__((ext_vector_type(8)))  float    v8f;

#define D_DIM 128
#define SPLIT 4   // column-sweep split factor per 16-row block (power of 2)

// ---------------------------------------------------------------------------
// Stage 1: row-normalize z = concat(z_i, z_j) -> f16 zn in workspace.
// One wave per row (32 lanes x 4 floats = 128). 8 waves per 256-thread block.
// ---------------------------------------------------------------------------
__global__ void ntxent_norm_kernel(const float* __restrict__ zi,
                                   const float* __restrict__ zj,
                                   _Float16* __restrict__ zn, int Bsz) {
  const int lane = threadIdx.x & 31;
  const int warp = threadIdx.x >> 5;
  const int row  = blockIdx.x * 8 + warp;

  const float* src = (row < Bsz) ? (zi + (size_t)row * D_DIM)
                                 : (zj + (size_t)(row - Bsz) * D_DIM);
  float4 v = *(const float4*)(src + lane * 4);
  float ss = v.x * v.x + v.y * v.y + v.z * v.z + v.w * v.w;
#pragma unroll
  for (int off = 1; off < 32; off <<= 1) ss += __shfl_xor(ss, off);

  const float inv = 1.0f / fmaxf(sqrtf(ss), 1e-8f);
  v4h h;
  h.x = (_Float16)(v.x * inv);
  h.y = (_Float16)(v.y * inv);
  h.z = (_Float16)(v.z * inv);
  h.w = (_Float16)(v.w * inv);
  *(v4h*)(zn + (size_t)row * D_DIM + lane * 4) = h;
}

// ---------------------------------------------------------------------------
// Stage 2: wave (rb, sp) owns 16 rows (rb) and column tiles t = sp, sp+SPLIT,...
// 4 chained v_wmma_f32_16x16x32_f16 per 16x16 tile (K=128); ping-pong B
// buffers with a fully peeled, branch-free steady-state loop (scalar control
// flow via readfirstlane). Branch-free fixed-max LSE: sumExp += exp(2c - 2).
// Split-0 waves recompute the diagonal tile (subtract its term) and the
// positive-pair tile (extract pos; loss is linear in pos -> scalar sum).
// ---------------------------------------------------------------------------
__global__ void ntxent_sim_kernel(const _Float16* __restrict__ zn,
                                  float* __restrict__ rowSum,
                                  float* __restrict__ posPart, int Bsz) {
  const int N    = 2 * Bsz;
  const int lane = threadIdx.x & 31;
  // wid is wave-uniform; make that explicit so all loop control is scalar.
  const int wid  = __builtin_amdgcn_readfirstlane(
      blockIdx.x * (blockDim.x >> 5) + (threadIdx.x >> 5));
  const int rb   = wid / SPLIT;           // 16-row block index (SGPR)
  const int sp   = wid & (SPLIT - 1);     // column split       (SGPR)
  const int m0   = rb * 16;
  const int hi   = lane >> 4;
  const int lm   = lane & 15;

  // --- A fragments (16 rows x 128 K), contiguous K order (same as B):
  // lane<16 : row M=lm, K = {kb+0..15}
  // lane>=16: row M=lm, K = {kb+16..31}
  const _Float16* arow = zn + (size_t)(m0 + lm) * D_DIM;
  v16h a[4];
#pragma unroll
  for (int kc = 0; kc < 4; ++kc) {
    const int kb = kc * 32;
    v8h lo = *(const v8h*)(arow + kb + hi * 16);
    v8h hh = *(const v8h*)(arow + kb + hi * 16 + 8);
    a[kc] = __builtin_shufflevector(lo, hh, 0, 1, 2, 3, 4, 5, 6, 7,
                                    8, 9, 10, 11, 12, 13, 14, 15);
  }

  float sumExp[8] = {0.f, 0.f, 0.f, 0.f, 0.f, 0.f, 0.f, 0.f};

  // B fragment, contiguous K order (same as A): lane holds 16 contiguous K-halves of
  // column t*16+lm; lanes 0-15 -> K kb+0..15, lanes 16-31 -> K kb+16..31.
  // Per-lane vector base; per-tile offset is scalar (t is SGPR).
  const _Float16* zrow = zn + (size_t)lm * D_DIM + hi * 16;
  auto loadB = [&](int t, v16h* b) {
    const _Float16* p = zrow + (size_t)t * (16 * D_DIM);
#pragma unroll
    for (int kc = 0; kc < 4; ++kc) b[kc] = *(const v16h*)(p + kc * 32);
  };
  auto mmTile = [&](const v16h* b) -> v8f {
    v8f c = {};
#pragma unroll
    for (int kc = 0; kc < 4; ++kc)
      c = __builtin_amdgcn_wmma_f32_16x16x32_f16(
          false, a[kc], false, b[kc], (short)0, c, false, false);
    return c;
  };
  auto accum = [&](const v16h* b) {
    v8f c = mmTile(b);
#pragma unroll
    for (int i = 0; i < 8; ++i)
      sumExp[i] += __expf(__builtin_fmaf(c[i], 2.0f, -2.0f));
  };

  const int nTiles       = N / 16;
  const int tilesPerWave = nTiles / SPLIT;  // 128 (even)

  v16h b0[4], b1[4];
  loadB(sp, b0);
  loadB(sp + SPLIT, b1);
  int t = sp;
  // Steady state: no branches; each load clause hides behind a full tile of
  // WMMA + exp work from the other buffer.
  for (int it = 0; it < tilesPerWave - 2; it += 2) {
    accum(b0);
    loadB(t + 2 * SPLIT, b0);
    accum(b1);
    loadB(t + 3 * SPLIT, b1);
    t += 2 * SPLIT;
  }
  accum(b0);  // drain
  accum(b1);

  // --- Corrections (scalar branch; EXEC stays all-ones for WMMA).
  float posSum = 0.f;
  if (sp == 0) {
    v16h bd[4];
    // Diagonal tile t == rb: element n==m iff lm == hi*8+i.
    loadB(rb, bd);
    v8f c = mmTile(bd);
#pragma unroll
    for (int i = 0; i < 8; ++i)
      if (lm == hi * 8 + i)
        sumExp[i] -= __expf(__builtin_fmaf(c[i], 2.0f, -2.0f));
    // Positive tile t == rb ^ (Bsz/16): n == m^Bsz iff lm == hi*8+i.
    loadB(rb ^ (Bsz >> 4), bd);
    c = mmTile(bd);
#pragma unroll
    for (int i = 0; i < 8; ++i)
      if (lm == hi * 8 + i) posSum += c[i] * 2.0f;
  }

  // Row partial sums: reduce across the 16 lanes of each half-wave.
#pragma unroll
  for (int i = 0; i < 8; ++i) {
    float s = sumExp[i];
    s += __shfl_xor(s, 1);
    s += __shfl_xor(s, 2);
    s += __shfl_xor(s, 4);
    s += __shfl_xor(s, 8);
    sumExp[i] = s;
  }
  if (lm == 0) {  // lane 0 -> rows m0+0..7, lane 16 -> rows m0+8..15
    const int base = m0 + hi * 8;
#pragma unroll
    for (int i = 0; i < 8; ++i)
      rowSum[(size_t)(base + i) * SPLIT + sp] = sumExp[i];
  }

  posSum += __shfl_xor(posSum, 1);
  posSum += __shfl_xor(posSum, 2);
  posSum += __shfl_xor(posSum, 4);
  posSum += __shfl_xor(posSum, 8);
  posSum += __shfl_xor(posSum, 16);
  if (lane == 0) posPart[wid] = posSum;  // zero for sp != 0 (always written)
}

// ---------------------------------------------------------------------------
// Stage 3: combine SPLIT partials per row, lse = 2 + log(sum), subtract pos,
// deterministic single-block reduction, /N.
// ---------------------------------------------------------------------------
__global__ void ntxent_finalize_kernel(const float* __restrict__ rowSum,
                                       const float* __restrict__ posPart,
                                       int N, int nWaves,
                                       float* __restrict__ out) {
  float acc = 0.f;
  for (int r = threadIdx.x; r < N; r += 256) {
    float s = 0.f;
#pragma unroll
    for (int k = 0; k < SPLIT; ++k) s += rowSum[(size_t)r * SPLIT + k];
    acc += 2.0f + __logf(s);
  }
  for (int w = threadIdx.x; w < nWaves; w += 256) acc -= posPart[w];

  __shared__ float red[256];
  red[threadIdx.x] = acc;
  __syncthreads();
  for (int s = 128; s > 0; s >>= 1) {
    if (threadIdx.x < s) red[threadIdx.x] += red[threadIdx.x + s];
    __syncthreads();
  }
  if (threadIdx.x == 0) out[0] = red[0] / (float)N;
}

extern "C" void kernel_launch(void* const* d_in, const int* in_sizes, int n_in,
                              void* d_out, int out_size, void* d_ws, size_t ws_size,
                              hipStream_t stream) {
  const float* zi = (const float*)d_in[0];
  const float* zj = (const float*)d_in[1];
  float* out = (float*)d_out;

  const int Bsz = in_sizes[0] / D_DIM;  // 4096
  const int N   = 2 * Bsz;              // 8192

  _Float16* zn = (_Float16*)d_ws;
  const size_t znBytes = (size_t)N * D_DIM * sizeof(_Float16);       // 2 MB
  float* rowSum  = (float*)((char*)d_ws + znBytes);                  // N*SPLIT
  float* posPart = rowSum + (size_t)N * SPLIT;                       // nWaves

  // Stage 1: 1 wave per row, 8 waves per block.
  ntxent_norm_kernel<<<N / 8, 256, 0, stream>>>(zi, zj, zn, Bsz);

  // Stage 2: (N/16)*SPLIT waves, 8 waves per block.
  const int nWaves = (N / 16) * SPLIT;
  ntxent_sim_kernel<<<nWaves / 8, 256, 0, stream>>>(zn, rowSum, posPart, Bsz);

  // Stage 3: single block.
  ntxent_finalize_kernel<<<1, 256, 0, stream>>>(rowSum, posPart, N, nWaves, out);
}
